// HFRM_66168266162255
// MI455X (gfx1250) — matrix operand order repaired and checked by reading, not compiled, not run on hardware
//
#include <hip/hip_runtime.h>
#include <hip/hip_bf16.h>

typedef __attribute__((ext_vector_type(16))) _Float16 v16h;
typedef __attribute__((ext_vector_type(8)))  float    v8f;
typedef _Float16 half_t;

#define WMMA_F16(a,b,c) __builtin_amdgcn_wmma_f32_16x16x32_f16( \
    false,(a),false,(b),(short)0,(c),false,false)

// ---- WMMA fragment index maps (contiguous k order chosen here, not the ISA table order; 16-bit, wave32) ----
// A 16x32 (MxK): lanes 0-15 M=lane; e=0..15 -> k=kh*16+e
// (one contiguous 16-f16 run)
__device__ __forceinline__ int amap(int lane, int e) {
    int kh = (lane >> 4) & 1;
    return kh * 16 + e;
}
// B 32x16 (KxN): lanes 0-15 K=0..15, lanes 16-31 K=16..31 (one 16-f16 run)
__device__ __forceinline__ int bmap(int lane, int e) {
    return (((lane >> 4) & 1) * 16) + e;
}

// ---- gfx1250 async global->LDS copy (ASYNCcnt-tracked DMA path) ----
// Per-lane 16B transfer: vdst = LDS byte offset VGPR, then 64-bit vaddr.
__device__ __forceinline__ void async_g2l_b128(void* lds, const void* g) {
    asm volatile("global_load_async_to_lds_b128 %0, %1, off"
                 :: "v"((unsigned)(unsigned long long)lds), "v"(g) : "memory");
}
__device__ __forceinline__ void wait_async0() {
    asm volatile("s_wait_asynccnt 0x0" ::: "memory");
}

// =====================================================================
// One-shot weight convert: src f32 [o][c][taps] -> dst f16 [tap][o*64+c]
// =====================================================================
__global__ void wcvt_kernel(const float* __restrict__ src,
                            half_t* __restrict__ dst, int taps)
{
    int idx = blockIdx.x * 256 + threadIdx.x;
    if (idx >= 4096 * taps) return;
    int tap = idx >> 12, oc = idx & 4095;
    dst[idx] = (half_t)src[(size_t)oc * taps + tap];
}

// =====================================================================
// Dense conv via WMMA implicit GEMM: out[b,o,y,x] =
//   act( bias[o] + sum_{tap,c} W[o,c,tap] * in[b,c,y+dy,x+dx] ) (+ res)
// taps = 9 (3x3 dilated) or 1 (pointwise). 64 in/out ch, H=W=256.
// Block 256 thr = 8 waves; WG = (b, 16-px x-strip, 8-row y-block);
// wave wv computes row y0+wv, 4 WMMA M-tiles (64 out ch).
// LDS: activations channel-contiguous [pixel][c], pitch 72 f16 (144B=9*16B,
// 16B-aligned fragments, conflict-free b128 banking); weights [o][c] pitch 72.
// Weight slabs are staged with async global->LDS b128 DMA (layout-preserving).
// =====================================================================
#define SP 72   // s_in pixel pitch (f16)
#define SW 72   // s_w  row  pitch (f16)

__global__ __launch_bounds__(256)
void conv64_wmma(const half_t* __restrict__ in, const half_t* __restrict__ wgt16,
                 const float* __restrict__ bias, const half_t* __restrict__ res,
                 half_t* __restrict__ out, int taps, int dil, int relu)
{
    const int H = 256, W = 256;
    __shared__ __align__(16) half_t s_in[14 * 22 * SP]; // max halo tile: 44.3 KB
    __shared__ __align__(16) half_t s_w[64 * SW];       // tap weights: 9.2 KB

    const int tid = threadIdx.x, lane = tid & 31, wv = tid >> 5;
    const int x0 = blockIdx.x * 16, y0 = blockIdx.y * 8, b = blockIdx.z;
    const int pad  = (taps == 9) ? dil : 0;
    const int ROWS = 8 + 2 * pad, XW = 16 + 2 * pad;
    const size_t plane = (size_t)H * W;
    const half_t* inb = in + (size_t)b * 64 * plane;

    // --- kick off async DMA of tap-0 weight slab into LDS ---
    {
        const half_t* wsrc = wgt16;
        for (int t = tid; t < 512; t += 256) {          // 512 x 16B chunks
            int o = t >> 3, ch = (t & 7) * 8;
            async_g2l_b128(&s_w[o * SW + ch], wsrc + o * 64 + ch);
        }
    }

    // --- stage input halo tile, transposed to [pixel][c] ---
    const int tot = 64 * ROWS * XW;
    for (int i = tid; i < tot; i += 256) {
        int c = i / (ROWS * XW), pix = i % (ROWS * XW);
        int ry = pix / XW, rx = pix % XW;
        int gy = y0 - pad + ry, gx = x0 - pad + rx;
        half_t v = (half_t)0.f;
        if ((unsigned)gy < (unsigned)H && (unsigned)gx < (unsigned)W)
            v = inb[(size_t)c * plane + (size_t)gy * W + gx];
        s_in[pix * SP + c] = v;
    }

    // --- init accumulators with bias (C layout: VGPR r -> M=r / r+8) ---
    v8f acc[4];
#pragma unroll
    for (int mt = 0; mt < 4; ++mt)
#pragma unroll
        for (int r = 0; r < 8; ++r) {
            int o = mt * 16 + ((lane < 16) ? r : r + 8);
            acc[mt][r] = bias[o];
        }
    wait_async0();
    __syncthreads();

    const int y  = y0 + wv;
    const int kh16 = ((lane >> 4) & 1) * 16;
    for (int tap = 0; tap < taps; ++tap) {
        int dy = (taps == 9) ? (tap / 3 - 1) * dil : 0;
        int dx = (taps == 9) ? (tap % 3 - 1) * dil : 0;
        if (tap + 1 < taps)   // gfx1250 global_prefetch_b8: next tap slab
            __builtin_prefetch(wgt16 + (size_t)(tap + 1) * 4096 + tid * 16, 0, 1);

        const int pixb = (wv + pad + dy) * XW + (lane & 15) + pad + dx;
#pragma unroll
        for (int ks = 0; ks < 2; ++ks) {
            const int c0 = ks * 32;
            v16h bf;
            const int bb = pixb * SP + c0 + kh16;
#pragma unroll
            for (int e = 0; e < 16; ++e) bf[e] = s_in[bb + e];
#pragma unroll
            for (int mt = 0; mt < 4; ++mt) {
                const int ab = (mt * 16 + (lane & 15)) * SW + c0;
                v16h af;
#pragma unroll
                for (int e = 0; e < 16; ++e) af[e] = s_w[ab + amap(lane, e)];
                acc[mt] = WMMA_F16(af, bf, acc[mt]);
            }
        }
        __syncthreads();
        // async-DMA next tap's weight slab over the consumed one
        if (tap + 1 < taps) {
            const half_t* wsrc = wgt16 + (size_t)(tap + 1) * 4096;
            for (int t = tid; t < 512; t += 256) {
                int o = t >> 3, ch = (t & 7) * 8;
                async_g2l_b128(&s_w[o * SW + ch], wsrc + o * 64 + ch);
            }
            wait_async0();
            __syncthreads();
        }
    }

    // --- epilogue: leaky-relu / residual add / f16 store ---
    const int xn = x0 + (lane & 15);
    half_t* outb = out + (size_t)b * 64 * plane;
    const half_t* resb = res ? res + (size_t)b * 64 * plane : nullptr;
#pragma unroll
    for (int mt = 0; mt < 4; ++mt)
#pragma unroll
        for (int r = 0; r < 8; ++r) {
            int o = mt * 16 + ((lane < 16) ? r : r + 8);
            float v = acc[mt][r];
            if (relu) v = (v > 0.f) ? v : 0.01f * v;
            size_t oi = (size_t)o * plane + (size_t)y * W + xn;
            if (resb) v += (float)resb[oi];
            outb[oi] = (half_t)v;
        }
}

// =====================================================================
// Per-row channel cross-attention via WMMA, one WG per (b, row h).
//   S[c,d] = (1/sqrt(8)) sum_w q[c,w] k[d,w]   (M=64,N=64,K=256)
//   P      = softmax_d(S);  O[c,w] = sum_d P[c,d] v[d,w]
// q/k chunk staging uses async global->LDS b128 DMA (layout-preserving);
// V is transposed into LDS so it uses the explicit path.
// =====================================================================
#define QS 40
#define SS 66
#define PS 72
#define VS 72

__global__ __launch_bounds__(256)
void attn_row_wmma(const half_t* __restrict__ q, const half_t* __restrict__ kk,
                   const half_t* __restrict__ vv, half_t* __restrict__ o)
{
    __shared__ __align__(16) float smemf[11776];   // 46 KB, region-reused
    half_t* qc = (half_t*)smemf;                   // [64][QS]       f16, 5120B
    half_t* kc = (half_t*)(smemf + 1280);          // [64][QS]       f16, 5120B
    float*  S  = (float*)(smemf + 2560);           // [64][SS]       f32, 16.9KB
    half_t* P  = (half_t*)smemf;                   // [64][PS]       f16, 9216B
    half_t* V  = (half_t*)(smemf + 2560);          // [256 w][VS d]  f16, 36.9KB

    const int tid = threadIdx.x, lane = tid & 31, wv = tid >> 5;
    const int b = blockIdx.y, h = blockIdx.x;
    const size_t rowbase = (((size_t)b * 64) * 256 + h) * 256;
    const int kh16 = ((lane >> 4) & 1) * 16;

    // ---- phase A: S = q * k^T over 8 chunks of K(w)=32 ----
    const int mt = wv >> 1, nh = wv & 1;
    v8f sacc[2];
#pragma unroll
    for (int j = 0; j < 2; ++j)
#pragma unroll
        for (int r = 0; r < 8; ++r) sacc[j][r] = 0.f;

    for (int wc = 0; wc < 8; ++wc) {
        // async-DMA q and k chunks: 2 buffers * 64 rows * 4 x 16B chunks
        for (int t = tid; t < 512; t += 256) {
            int buf = t >> 8, i = t & 255;
            int c = i >> 2, ch = (i & 3) * 8;
            const half_t* g = (buf ? kk : q) + rowbase + (size_t)c * 65536
                              + wc * 32 + ch;
            half_t* l = (buf ? kc : qc) + c * QS + ch;
            async_g2l_b128(l, g);
        }
        wait_async0();
        __syncthreads();
        v16h af;
        const int ab = (mt * 16 + (lane & 15)) * QS;
#pragma unroll
        for (int e = 0; e < 16; ++e) af[e] = qc[ab + amap(lane, e)];
#pragma unroll
        for (int j = 0; j < 2; ++j) {
            const int bb = ((nh * 2 + j) * 16 + (lane & 15)) * QS + kh16;
            v16h bf;
#pragma unroll
            for (int e = 0; e < 16; ++e) bf[e] = kc[bb + e];
            sacc[j] = WMMA_F16(af, bf, sacc[j]);
        }
        __syncthreads();
    }
    const float scale = 0.35355339059327373f;      // 1/sqrt(head_size=8)
#pragma unroll
    for (int j = 0; j < 2; ++j)
#pragma unroll
        for (int r = 0; r < 8; ++r) {
            int c = mt * 16 + ((lane < 16) ? r : r + 8);
            int d = (nh * 2 + j) * 16 + (lane & 15);
            S[c * SS + d] = sacc[j][r] * scale;
        }
    __syncthreads();

    // ---- softmax over d (64 lanes, one row each) ----
    if (tid < 64) {
        int c = tid;
        float mx = -1e30f;
        for (int d = 0; d < 64; ++d) mx = fmaxf(mx, S[c * SS + d]);
        float sum = 0.f;
        for (int d = 0; d < 64; ++d) {
            float e = __expf(S[c * SS + d] - mx);
            S[c * SS + d] = e;
            sum += e;
        }
        float inv = 1.f / sum;
        for (int d = 0; d < 64; ++d)
            P[c * PS + d] = (half_t)(S[c * SS + d] * inv);
    }
    __syncthreads();

    // ---- phase B: O = P * v (V staged transposed [w][d]) ----
    for (int i = tid; i < 16384; i += 256) {
        int d = i >> 8, w = i & 255;
        V[w * VS + d] = vv[rowbase + (size_t)d * 65536 + w];
    }
    __syncthreads();

    const int mt2 = wv & 3, hf = wv >> 2;
    v8f oacc[8];
#pragma unroll
    for (int i = 0; i < 8; ++i)
#pragma unroll
        for (int r = 0; r < 8; ++r) oacc[i][r] = 0.f;

#pragma unroll
    for (int ks = 0; ks < 2; ++ks) {
        const int c0 = ks * 32;
        v16h af;
        const int ab = (mt2 * 16 + (lane & 15)) * PS + c0;
#pragma unroll
        for (int e = 0; e < 16; ++e) af[e] = P[ab + amap(lane, e)];
#pragma unroll
        for (int i = 0; i < 8; ++i) {
            const int w = (hf * 8 + i) * 16 + (lane & 15);
            const int bb = w * VS + c0 + kh16;
            v16h bf;
#pragma unroll
            for (int e = 0; e < 16; ++e) bf[e] = V[bb + e];
            oacc[i] = WMMA_F16(af, bf, oacc[i]);
        }
    }
#pragma unroll
    for (int i = 0; i < 8; ++i)
#pragma unroll
        for (int r = 0; r < 8; ++r) {
            int c = mt2 * 16 + ((lane < 16) ? r : r + 8);
            int w = (hf * 8 + i) * 16 + (lane & 15);
            o[rowbase + (size_t)c * 65536 + w] = (half_t)oacc[i][r];
        }
}

// =====================================================================
// Small VALU kernels: heads (3->64), depthwise 3x3 (64ch), tail, out.
// =====================================================================
__global__ void head_x_kernel(const float* __restrict__ x,
                              const float* __restrict__ dww, const float* __restrict__ dwb,
                              const float* __restrict__ pww, const float* __restrict__ pwb,
                              half_t* __restrict__ out)
{
    int idx = blockIdx.x * 256 + threadIdx.x;
    if (idx >= 6 * 65536) return;
    int b = idx >> 16, p = idx & 65535, yy = p >> 8, xx = p & 255;
    float hv[3];
    for (int c = 0; c < 3; ++c) {
        float a = dwb[c];
        for (int t = 0; t < 9; ++t) {
            int gy = yy + t / 3 - 1, gx = xx + t % 3 - 1;
            if ((unsigned)gy < 256u && (unsigned)gx < 256u)
                a += x[((size_t)(b * 3 + c) << 16) + (gy << 8) + gx] * dww[c * 9 + t];
        }
        hv[c] = a;
    }
    for (int o = 0; o < 64; ++o) {
        float v = pwb[o] + pww[o * 3] * hv[0] + pww[o * 3 + 1] * hv[1] + pww[o * 3 + 2] * hv[2];
        out[(((size_t)b * 64 + o) << 16) + p] = (half_t)v;
    }
}

__device__ __forceinline__ float bsample(const float* y, int b, int c, int gy, int gx)
{
    // jax.image.resize bilinear, 128 -> 256, half-pixel centers, edge clamp
    float fy = gy * 0.5f - 0.25f, fx = gx * 0.5f - 0.25f;
    fy = fminf(fmaxf(fy, 0.f), 127.f);
    fx = fminf(fmaxf(fx, 0.f), 127.f);
    int y0 = (int)floorf(fy), x0 = (int)floorf(fx);
    int y1 = (y0 + 1 < 127) ? y0 + 1 : 127;
    int x1 = (x0 + 1 < 127) ? x0 + 1 : 127;
    float wy = fy - (float)y0, wx = fx - (float)x0;
    const float* pl = y + ((size_t)(b * 3 + c) << 14);
    float v00 = pl[(y0 << 7) + x0], v01 = pl[(y0 << 7) + x1];
    float v10 = pl[(y1 << 7) + x0], v11 = pl[(y1 << 7) + x1];
    return v00 * (1.f - wy) * (1.f - wx) + v01 * (1.f - wy) * wx
         + v10 * wy * (1.f - wx) + v11 * wy * wx;
}

__global__ void head_y_kernel(const float* __restrict__ y,
                              const float* __restrict__ dww, const float* __restrict__ dwb,
                              const float* __restrict__ pww, const float* __restrict__ pwb,
                              half_t* __restrict__ out)
{
    int idx = blockIdx.x * 256 + threadIdx.x;
    if (idx >= 6 * 65536) return;
    int b = idx >> 16, p = idx & 65535, yy = p >> 8, xx = p & 255;
    float hv[3];
    for (int c = 0; c < 3; ++c) {
        float a = dwb[c];
        for (int t = 0; t < 9; ++t) {
            int gy = yy + t / 3 - 1, gx = xx + t % 3 - 1;
            if ((unsigned)gy < 256u && (unsigned)gx < 256u)
                a += bsample(y, b, c, gy, gx) * dww[c * 9 + t];
        }
        hv[c] = a;
    }
    for (int o = 0; o < 64; ++o) {
        float v = pwb[o] + pww[o * 3] * hv[0] + pww[o * 3 + 1] * hv[1] + pww[o * 3 + 2] * hv[2];
        out[(((size_t)b * 64 + o) << 16) + p] = (half_t)v;
    }
}

__global__ void dw64_kernel(const half_t* __restrict__ in,
                            const float* __restrict__ dww, const float* __restrict__ dwb,
                            half_t* __restrict__ out)
{
    int idx = blockIdx.x * 256 + threadIdx.x;     // 2*64*65536
    if (idx >= 8388608) return;
    int p = idx & 65535, cc = (idx >> 16) & 63;
    int yy = p >> 8, xx = p & 255;
    size_t base = (size_t)(idx >> 16) << 16;      // (b*64+cc) plane base
    float a = dwb[cc];
    for (int t = 0; t < 9; ++t) {
        int gy = yy + t / 3 - 1, gx = xx + t % 3 - 1;
        if ((unsigned)gy < 256u && (unsigned)gx < 256u)
            a += (float)in[base + (gy << 8) + gx] * dww[cc * 9 + t];
    }
    out[base + p] = (half_t)a;
}

__global__ void tail_kernel(const half_t* __restrict__ in,
                            const float* __restrict__ dww, const float* __restrict__ dwb,
                            const float* __restrict__ pww, const float* __restrict__ pwb,
                            const float* __restrict__ xres, float* __restrict__ out)
{
    int idx = blockIdx.x * 256 + threadIdx.x;
    if (idx >= 6 * 65536) return;
    int b = idx >> 16, p = idx & 65535, yy = p >> 8, xx = p & 255;
    float acc[3];
    for (int o = 0; o < 3; ++o) acc[o] = pwb[o];
    for (int c = 0; c < 64; ++c) {
        float a = dwb[c];
        size_t base = (size_t)(b * 64 + c) << 16;
        for (int t = 0; t < 9; ++t) {
            int gy = yy + t / 3 - 1, gx = xx + t % 3 - 1;
            if ((unsigned)gy < 256u && (unsigned)gx < 256u)
                a += (float)in[base + (gy << 8) + gx] * dww[c * 9 + t];
        }
        for (int o = 0; o < 3; ++o) acc[o] += pww[o * 64 + c] * a;
    }
    for (int o = 0; o < 3; ++o)
        out[((size_t)(b * 3 + o) << 16) + p] =
            acc[o] + xres[((size_t)(b * 3 + o) << 16) + p];
}

__global__ void out_kernel(const float* __restrict__ in,
                           const float* __restrict__ dww, const float* __restrict__ dwb,
                           const float* __restrict__ pww, const float* __restrict__ pwb,
                           float* __restrict__ out)
{
    int idx = blockIdx.x * 256 + threadIdx.x;
    if (idx >= 6 * 65536) return;
    int b = idx >> 16, p = idx & 65535, yy = p >> 8, xx = p & 255;
    float hv[3];
    for (int c = 0; c < 3; ++c) {
        float a = dwb[c];
        for (int t = 0; t < 9; ++t) {
            int gy = yy + t / 3 - 1, gx = xx + t % 3 - 1;
            if ((unsigned)gy < 256u && (unsigned)gx < 256u)
                a += in[((size_t)(b * 3 + c) << 16) + (gy << 8) + gx] * dww[c * 9 + t];
        }
        hv[c] = a;
    }
    for (int o = 0; o < 3; ++o)
        out[((size_t)(b * 3 + o) << 16) + p] =
            pwb[o] + pww[o * 3] * hv[0] + pww[o * 3 + 1] * hv[1] + pww[o * 3 + 2] * hv[2];
}

// =====================================================================
// Host orchestration.
// d_in order (setup_inputs insertion order, nested dicts flattened):
//  0:x 1:y | head: 2..5 | attn0: 6..17 | attn1: 18..29 | attn2: 30..41 |
//  res_HL: 42..51 | res_LH: 52..61 | res_HH: 62..71 | tail: 72..75 |
//  out: 76..79 | layer: 80
// Band mapping: band0=HL uses attn1+res_HL; band1=LH uses attn2+res_LH;
//               band2=HH uses attn0+res_HH.
// =====================================================================
extern "C" void kernel_launch(void* const* d_in, const int* in_sizes, int n_in,
                              void* d_out, int out_size, void* d_ws, size_t ws_size,
                              hipStream_t stream)
{
    (void)in_sizes; (void)n_in; (void)out_size; (void)ws_size;
    const float* x = (const float*)d_in[0];
    const float* y = (const float*)d_in[1];
    auto P = [&](int i) { return (const float*)d_in[i]; };

    const size_t FULL = (size_t)6 * 64 * 65536;
    const size_t BE   = (size_t)2 * 64 * 65536;
    half_t* XH  = (half_t*)d_ws;
    half_t* YH  = XH + FULL;                      // also holds concat output
    half_t* QB  = YH + FULL;
    half_t* KB  = QB + BE;
    half_t* VB  = KB + BE;
    half_t* AB  = VB + BE;                        // attention out / res input
    half_t* Pg0 = AB + BE;
    half_t* Pg1 = Pg0 + BE;
    half_t* DWT = Pg1 + BE;                       // depthwise intermediate
    float*  T3  = (float*)(DWT + BE);             // (6,3,256,256) f32
    half_t* WK  = (half_t*)(T3 + (size_t)6 * 3 * 65536);  // f16 weight slot

    head_x_kernel<<<1536, 256, 0, stream>>>(x, P(2), P(3), P(4), P(5), XH);
    head_y_kernel<<<1536, 256, 0, stream>>>(y, P(2), P(3), P(4), P(5), YH);

    dim3 cgrid(16, 32, 2), cblk(256);
    // stream-ordered: convert weights into WK, then run the conv using WK
    auto conv = [&](const half_t* src, int wi, int bi, const half_t* res,
                    half_t* dst, int taps, int dil, int relu) {
        wcvt_kernel<<<(4096 * taps + 255) / 256, 256, 0, stream>>>(P(wi), WK, taps);
        conv64_wmma<<<cgrid, cblk, 0, stream>>>(src, WK, P(bi), res, dst,
                                                taps, dil, relu);
    };

    for (int band = 0; band < 3; ++band) {
        const int ab = (band == 0) ? 18 : (band == 1) ? 30 : 6;  // attn base
        const int rb = (band == 0) ? 42 : (band == 1) ? 52 : 62; // res base
        half_t* xb = XH + (size_t)band * BE;      // ctx (k,v source)
        half_t* yb = YH + (size_t)band * BE;      // hidden (q source)

        dw64_kernel<<<32768, 256, 0, stream>>>(yb, P(ab + 0), P(ab + 1), DWT);
        conv(DWT, ab + 2, ab + 3, nullptr, QB, 1, 1, 0);
        dw64_kernel<<<32768, 256, 0, stream>>>(xb, P(ab + 4), P(ab + 5), DWT);
        conv(DWT, ab + 6, ab + 7, nullptr, KB, 1, 1, 0);
        dw64_kernel<<<32768, 256, 0, stream>>>(xb, P(ab + 8), P(ab + 9), DWT);
        conv(DWT, ab + 10, ab + 11, nullptr, VB, 1, 1, 0);

        attn_row_wmma<<<dim3(256, 2), 256, 0, stream>>>(QB, KB, VB, AB);

        // dilated resblock: dils (1,2,3,2) with leaky-relu, then 3x3 + residual
        conv(AB,  rb + 0, rb + 1, nullptr, Pg0, 9, 1, 1);
        conv(Pg0, rb + 2, rb + 3, nullptr, Pg1, 9, 2, 1);
        conv(Pg1, rb + 4, rb + 5, nullptr, Pg0, 9, 3, 1);
        conv(Pg0, rb + 6, rb + 7, nullptr, Pg1, 9, 2, 1);
        conv(Pg1, rb + 8, rb + 9, AB, yb, 9, 1, 0);
    }

    tail_kernel<<<1536, 256, 0, stream>>>(YH, P(72), P(73), P(74), P(75), x, T3);
    out_kernel<<<1536, 256, 0, stream>>>(T3, P(76), P(77), P(78), P(79), (float*)d_out);
}
